// SimulatedBifurcationModel_6090263626261
// MI455X (gfx1250) — matrix operand order repaired and checked by reading, not compiled, not run on hardware
//
#include <hip/hip_runtime.h>

// ---------------------------------------------------------------------------
// Simulated Bifurcation, N=4096, 1000 sequential iterations.
//
// MI455X strategy:
//  * J converted once to bf16 (32 MB) -> resident in 192 MB L2; each iteration
//    is ONE fused pass J @ [x, sol] via v_wmma_f32_16x16x32_bf16.
//  * Persistent kernel (32 blocks x 256 thr, all co-resident) with a software
//    device barrier: 1 barrier/iteration instead of ~3 kernel launches.
//  * Block b owns rows [b*128, b*128+128): its WMMA output Jx feeds its own
//    update -> Jx/x/y stay in LDS; only packed bf16 x/sol go through global.
//  * B fragments (x / sol / zero columns) come from LDS -> no EXEC-mask
//    divergence in the inner loop: 2x global_load_b128 (J) + 2x ds_load_b128
//    (B) + 1 WMMA per 32-wide K chunk, unrolled x2 with two accumulators.
// ---------------------------------------------------------------------------

typedef __attribute__((ext_vector_type(16))) __bf16 v16bf;
typedef __attribute__((ext_vector_type(8)))  float  v8f;

#define SB_DT    0.25f
#define SB_NMAX  4096

// ---- bf16 helpers (round-to-nearest-even) ----------------------------------
__device__ __forceinline__ unsigned short f2bf(float f) {
    unsigned int u = __float_as_uint(f);
    u += 0x7FFFu + ((u >> 16) & 1u);
    return (unsigned short)(u >> 16);
}
__device__ __forceinline__ float bf2f(unsigned short b) {
    return __uint_as_float(((unsigned int)b) << 16);
}
__device__ __forceinline__ unsigned int pack2bf(float a, float b) {
    return (unsigned int)f2bf(a) | ((unsigned int)f2bf(b) << 16);
}

// ---------------------------------------------------------------------------
// Kernel 1: J (f32) -> Jb (bf16) + per-block partial sums of J.
// ---------------------------------------------------------------------------
__global__ __launch_bounds__(256)
void sb_convert(const float4* __restrict__ J4, uint2* __restrict__ Jb2,
                float* __restrict__ partials, long nf4) {
    float s = 0.0f;
    long stride = (long)gridDim.x * blockDim.x;
    for (long i = (long)blockIdx.x * blockDim.x + threadIdx.x; i < nf4; i += stride) {
        float4 v = J4[i];
        s += (v.x + v.y) + (v.z + v.w);
        uint2 o;
        o.x = pack2bf(v.x, v.y);
        o.y = pack2bf(v.z, v.w);
        Jb2[i] = o;
    }
    __shared__ float sm[256];
    sm[threadIdx.x] = s;
    __syncthreads();
    for (int off = 128; off > 0; off >>= 1) {
        if ((int)threadIdx.x < off) sm[threadIdx.x] += sm[threadIdx.x + off];
        __syncthreads();
    }
    if (threadIdx.x == 0) partials[blockIdx.x] = sm[0];
}

__global__ __launch_bounds__(256)
void sb_jsum_final(const float* __restrict__ partials, float* __restrict__ Jsum, int n) {
    float s = 0.0f;
    for (int i = threadIdx.x; i < n; i += 256) s += partials[i];
    __shared__ float sm[256];
    sm[threadIdx.x] = s;
    __syncthreads();
    for (int off = 128; off > 0; off >>= 1) {
        if ((int)threadIdx.x < off) sm[threadIdx.x] += sm[threadIdx.x + off];
        __syncthreads();
    }
    if (threadIdx.x == 0) Jsum[0] = sm[0];
}

// ---------------------------------------------------------------------------
// Kernel 2: init packed vectors (xb = bf16(init_x), sb = 0) + barrier state.
// ---------------------------------------------------------------------------
__global__ __launch_bounds__(256)
void sb_pack0(const float* __restrict__ x0,
              unsigned short* __restrict__ xb, unsigned short* __restrict__ sb,
              unsigned int* __restrict__ cnt, unsigned int* __restrict__ gen, int N) {
    int i = blockIdx.x * blockDim.x + threadIdx.x;
    if (i < N) {
        xb[i] = f2bf(x0[i]);
        sb[i] = 0;
    }
    if (i == 0) { cnt[0] = 0u; gen[0] = 0u; }
}

// ---------------------------------------------------------------------------
// Device-wide software barrier (sense via generation counter).
// ---------------------------------------------------------------------------
__device__ __forceinline__ void grid_barrier(unsigned int* cnt, unsigned int* gen,
                                             unsigned int nblocks) {
    __threadfence();                 // each wave drains its own global stores
    __syncthreads();
    if (threadIdx.x == 0) {
        unsigned int g = __hip_atomic_load(gen, __ATOMIC_RELAXED, __HIP_MEMORY_SCOPE_AGENT);
        unsigned int a = __hip_atomic_fetch_add(cnt, 1u, __ATOMIC_ACQ_REL, __HIP_MEMORY_SCOPE_AGENT);
        if (a == nblocks - 1u) {
            __hip_atomic_store(cnt, 0u, __ATOMIC_RELAXED, __HIP_MEMORY_SCOPE_AGENT);
            __hip_atomic_fetch_add(gen, 1u, __ATOMIC_RELEASE, __HIP_MEMORY_SCOPE_AGENT);
        } else {
            while (__hip_atomic_load(gen, __ATOMIC_ACQUIRE, __HIP_MEMORY_SCOPE_AGENT) == g) {
                __builtin_amdgcn_s_sleep(2);
            }
        }
    }
    __syncthreads();
}

// ---------------------------------------------------------------------------
// One WMMA step: A tile (16x32 bf16 of J, global) x B tile (32x16 bf16, LDS).
// A layout: lane m=row; lanes 0-15 K=k0+0..7 & k0+16..23, lanes 16-31
// K=k0+8..15 & k0+24..31 (two aligned 16B loads).
// B layout: lane L: N=L%16, same K split as A (two aligned 16B loads).
// ---------------------------------------------------------------------------
__device__ __forceinline__ v8f wmma_step(const unsigned short* __restrict__ ap,
                                         const unsigned short* bp, v8f c) {
    union { uint4 u[2]; v16bf v; } a, b;
    a.u[0] = *(const uint4*)(ap);
    a.u[1] = *(const uint4*)(ap + 16);
    b.u[0] = *(const uint4*)(bp);
    b.u[1] = *(const uint4*)(bp + 16);
    return __builtin_amdgcn_wmma_f32_16x16x32_bf16(
        false, a.v, false, b.v, (short)0, c, false, false);
}

// ---------------------------------------------------------------------------
// Kernel 3: persistent solver. Grid = N/128 blocks x 256 threads.
// Per iteration u (u = 0..iters):
//   matvec phase: copy xb,sb -> LDS; WMMA sweep over own 128 rows;
//     Jx -> LDS, dot(sol,Js) -> dot_part[u&1][b];
//     block 0 finalizes energies[u-2] from dot_part[(u-1)&1][*].
//   update phase (u < iters): block-local x,y update; publish xb,sb.
//   grid barrier.
// After loop: block 0 finalizes energies[iters-1].
// ---------------------------------------------------------------------------
__global__ __launch_bounds__(256)
void sb_persistent(const unsigned short* __restrict__ Jb,
                   unsigned short* __restrict__ xb, unsigned short* __restrict__ sb,
                   const float* __restrict__ x0, const float* __restrict__ y0,
                   float* __restrict__ dot_part, const float* __restrict__ Jsum,
                   float* __restrict__ energies,
                   unsigned int* cnt, unsigned int* gen,
                   int N, int iters, float xi) {
    __shared__ alignas(16) unsigned short vls[3 * SB_NMAX]; // [x | sol | zeros] bf16
    __shared__ float Jxls[128];
    __shared__ float xls[128];
    __shared__ float yls[128];
    __shared__ float wred[16];

    const int tid  = threadIdx.x;
    const int lane = tid & 31;
    const int wave = tid >> 5;
    const int nblocks = gridDim.x;
    const int per  = N / nblocks;                 // 128 rows/elements per block
    const int i0   = blockIdx.x * per;

    // --- one-time LDS init: zero column region, load local x/y state -------
    for (int i = tid; i < N; i += blockDim.x) vls[2 * N + i] = 0;
    for (int j = tid; j < per; j += blockDim.x) {
        xls[j] = x0[i0 + j];
        yls[j] = y0[i0 + j];
    }

    // --- WMMA geometry (fixed per thread) -----------------------------------
    const int half = lane >> 4;
    const int m    = lane & 15;                   // A row in tile / D column
    const int rowl = wave * 16 + half * 8;        // local row base for D writes
    const unsigned short* Arow = Jb + (size_t)(i0 + wave * 16 + m) * (size_t)N
                                    + half * 8;
    const int sel = (m == 0) ? 0 : ((m == 1) ? 1 : 2);
    const unsigned short* Bls = vls + sel * N + half * 8;

    const int nvec4 = N / 8;                      // 16B chunks per bf16 vector
    uint4* xl4 = (uint4*)vls;
    uint4* sl4 = (uint4*)(vls + N);
    const float denom = (iters > 1) ? (float)(iters - 1) : 1.0f;

    for (int u = 0; u <= iters; ++u) {
        // ---- stage packed x / sol columns into LDS -------------------------
        const uint4* xg4 = (const uint4*)xb;
        const uint4* sg4 = (const uint4*)sb;
        for (int i = tid; i < nvec4; i += blockDim.x) {
            xl4[i] = xg4[i];
            sl4[i] = sg4[i];
        }
        // ---- overlapped: block 0 finalizes lag-2 energy --------------------
        if (blockIdx.x == 0 && tid == 0 && u >= 2) {
            const float* dp = dot_part + ((u - 1) & 1) * nblocks;
            float d = 0.0f;
            for (int b = 0; b < nblocks; ++b) d += dp[b];
            energies[u - 2] = -0.25f * Jsum[0] + 0.25f * d;
        }
        __syncthreads();

        // ---- WMMA sweep: [Jx, Js] = J_rows @ [x, sol] ----------------------
        v8f c0 = {0.f, 0.f, 0.f, 0.f, 0.f, 0.f, 0.f, 0.f};
        v8f c1 = {0.f, 0.f, 0.f, 0.f, 0.f, 0.f, 0.f, 0.f};
        for (int kk = 0; kk < N; kk += 64) {
            c0 = wmma_step(Arow + kk,      Bls + kk,      c0);
            c1 = wmma_step(Arow + kk + 32, Bls + kk + 32, c1);
        }
        v8f c = c0 + c1;

        if (m == 0) {                              // column 0 = Jx (stay local)
#pragma unroll
            for (int r = 0; r < 8; ++r) Jxls[rowl + r] = c[r];
        }
        if (m == 1) {                              // column 1 = Js -> dot w/ sol
            float p = 0.0f;
#pragma unroll
            for (int r = 0; r < 8; ++r) p += c[r] * bf2f(vls[N + i0 + rowl + r]);
            wred[wave * 2 + half] = p;
        }
        __syncthreads();
        if (tid == 0) {
            float d = 0.0f;
#pragma unroll
            for (int k = 0; k < 16; ++k) d += wred[k];
            dot_part[(u & 1) * nblocks + blockIdx.x] = d;
        }
        __syncthreads();

        // ---- block-local state update + publish bf16 x / sol ---------------
        if (u < iters) {
            float a = (float)u / denom;            // linspace(0,1,iters)[u]
            for (int j = tid; j < per; j += blockDim.x) {
                float xv = xls[j];
                float yv = yls[j];
                yv += ((a - 1.0f) * xv + xi * Jxls[j]) * SB_DT;
                xv += yv * SB_DT;
                if (fabsf(xv) > 1.0f) {
                    yv = 0.0f;
                    xv = (xv > 0.0f) ? 1.0f : -1.0f;
                }
                xls[j] = xv;
                yls[j] = yv;
                float s = (xv > 0.0f) ? 1.0f : ((xv < 0.0f) ? -1.0f : 0.0f);
                xb[i0 + j] = f2bf(xv);
                sb[i0 + j] = f2bf(s);
            }
        }
        grid_barrier(cnt, gen, (unsigned int)nblocks);
    }

    // ---- last energy (from matvec pass u == iters) -------------------------
    if (blockIdx.x == 0 && tid == 0) {
        const float* dp = dot_part + (iters & 1) * nblocks;
        float d = 0.0f;
        for (int b = 0; b < nblocks; ++b) d += dp[b];
        energies[iters - 1] = -0.25f * Jsum[0] + 0.25f * d;
    }
}

// ---------------------------------------------------------------------------
// Host orchestration: 4 launches total.
// Inputs: J [N*N] f32, init_x [N] f32, init_y [N] f32, num_iters (device
// scalar; host-side iteration count taken from out_size, which equals it).
// ---------------------------------------------------------------------------
extern "C" void kernel_launch(void* const* d_in, const int* in_sizes, int n_in,
                              void* d_out, int out_size, void* d_ws, size_t ws_size,
                              hipStream_t stream) {
    const float* J  = (const float*)d_in[0];
    const float* x0 = (const float*)d_in[1];
    const float* y0 = (const float*)d_in[2];
    float* energies = (float*)d_out;

    const int N = in_sizes[1];                     // 4096
    const int iters = (out_size > 0) ? out_size : 1000;
    const float xi = 0.7f / sqrtf((float)N);
    const int nblocks = N / 128;                   // 32: 8 waves x 16 rows each

    // Workspace layout (16B-aligned offsets).
    char* ws = (char*)d_ws;
    size_t off = 0;
    unsigned short* Jb = (unsigned short*)(ws + off); off += (size_t)N * N * 2; // 32 MB
    unsigned short* xb = (unsigned short*)(ws + off); off += (size_t)N * 2;
    unsigned short* sb = (unsigned short*)(ws + off); off += (size_t)N * 2;
    float* dot_part = (float*)(ws + off); off += (size_t)2 * nblocks * 4;
    const int CONV_BLOCKS = 4096;
    float* partials = (float*)(ws + off); off += (size_t)CONV_BLOCKS * 4;
    float* Jsum = (float*)(ws + off); off += 16;
    unsigned int* cnt = (unsigned int*)(ws + off); off += 16;
    unsigned int* gen = (unsigned int*)(ws + off); off += 16;
    (void)ws_size; (void)n_in;

    const long nf4 = (long)N * N / 4;
    sb_convert<<<CONV_BLOCKS, 256, 0, stream>>>((const float4*)J, (uint2*)Jb, partials, nf4);
    sb_jsum_final<<<1, 256, 0, stream>>>(partials, Jsum, CONV_BLOCKS);
    sb_pack0<<<(N + 255) / 256, 256, 0, stream>>>(x0, xb, sb, cnt, gen, N);
    sb_persistent<<<nblocks, 256, 0, stream>>>(Jb, xb, sb, x0, y0, dot_part,
                                               Jsum, energies, cnt, gen,
                                               N, iters, xi);
}
